// MiMoAudioAttention_37074157699362
// MI455X (gfx1250) — compile-verified, matrix operand order checked by reading, not run on hardware
//
#include <hip/hip_runtime.h>

#define HIDDEN 1024
#define NQ 16
#define NKV 4
#define HD 64
#define BATCH 2
#define SEQ 2048
#define TOKENS (BATCH * SEQ)
#define HALF (HD / 2)

typedef __attribute__((ext_vector_type(16))) __bf16 v16bf;
typedef __attribute__((ext_vector_type(4)))  __bf16 v4bf;
typedef __attribute__((ext_vector_type(8)))  float  v8f;

// K-offset pattern for 16-bit A/B fragments (ISA 7.12.2, 16-bit A 16x32):
// lanes 0-15 hold K {0..7, 16..23}, lanes 16-31 hold K {8..15, 24..31},
// two consecutive K per VGPR r.
__device__ __forceinline__ int frag_kbase(int r, int half) {
    return ((r >> 2) << 4) + (half << 3) + ((r & 3) << 1);
}

// Packed 4x f32 -> 4x bf16 8-byte LDS store (dest must be 8B aligned).
__device__ __forceinline__ void st_bf16x4(__bf16* p, float4 v) {
    v4bf t;
    t[0] = (__bf16)v.x; t[1] = (__bf16)v.y;
    t[2] = (__bf16)v.z; t[3] = (__bf16)v.w;
    *reinterpret_cast<v4bf*>(p) = t;
}

// ---------------------------------------------------------------------------
// Tiled GEMM: C[M,N] = A[M,K] @ W[K,N] (+ bias). fp32 in/out, bf16 WMMA math.
// Block: 128 threads (4 waves). Block tile 64x64; wave tile 32x32.
// W tile is stored TRANSPOSED in LDS so B-fragment reads are contiguous.
// Assumes M%64==0, N%64==0, K%32==0 (true for all launches here).
// ---------------------------------------------------------------------------
__global__ __launch_bounds__(128)
void gemm_bias_bf16(const float* __restrict__ A, const float* __restrict__ W,
                    const float* __restrict__ bias, float* __restrict__ C,
                    int M, int N, int K) {
    __shared__ __bf16 As[64][40];   // [m][k]  64x32 (+pad, 80B row stride)
    __shared__ __bf16 Wt[64][40];   // [n][k]  64x32 transposed (+pad)

    const int tid  = threadIdx.x;
    const int wave = tid >> 5, lane = tid & 31;
    const int half = lane >> 4, l16 = lane & 15;
    const int m0 = blockIdx.y << 6, n0 = blockIdx.x << 6;
    const int wm = (wave >> 1) << 5, wn = (wave & 1) << 5;

    v8f acc[2][2] = {};

    for (int kk = 0; kk < K; kk += 32) {
        // A tile: float4 global loads, packed bf16x4 LDS stores (contiguous).
        for (int i = tid; i < (64 * 32) / 4; i += 128) {
            int idx = i << 2;
            int r = idx >> 5, c = idx & 31;
            float4 a = *reinterpret_cast<const float4*>(
                &A[(size_t)(m0 + r) * K + kk + c]);
            st_bf16x4(&As[r][c], a);
        }
        // W tile: float4 global loads along N, transposed element stores.
        for (int i = tid; i < (32 * 64) / 4; i += 128) {
            int idx = i << 2;
            int r = idx >> 6, c = idx & 63;
            float4 w = *reinterpret_cast<const float4*>(
                &W[(size_t)(kk + r) * N + n0 + c]);
            Wt[c + 0][r] = (__bf16)w.x;
            Wt[c + 1][r] = (__bf16)w.y;
            Wt[c + 2][r] = (__bf16)w.z;
            Wt[c + 3][r] = (__bf16)w.w;
        }
        __syncthreads();

        v16bf af[2], bfg[2];
#pragma unroll
        for (int mi = 0; mi < 2; ++mi)
#pragma unroll
            for (int r = 0; r < 8; ++r) {
                int kb = frag_kbase(r, half);
                af[mi][2 * r]     = As[wm + mi * 16 + l16][kb];
                af[mi][2 * r + 1] = As[wm + mi * 16 + l16][kb + 1];
            }
#pragma unroll
        for (int ni = 0; ni < 2; ++ni)
#pragma unroll
            for (int r = 0; r < 8; ++r) {
                int kb = frag_kbase(r, half);
                bfg[ni][2 * r]     = Wt[wn + ni * 16 + l16][kb];
                bfg[ni][2 * r + 1] = Wt[wn + ni * 16 + l16][kb + 1];
            }
#pragma unroll
        for (int mi = 0; mi < 2; ++mi)
#pragma unroll
            for (int ni = 0; ni < 2; ++ni)
                acc[mi][ni] = __builtin_amdgcn_wmma_f32_16x16x32_bf16(
                    false, af[mi], false, bfg[ni], (short)0, acc[mi][ni],
                    false, false);
        __syncthreads();
    }

#pragma unroll
    for (int mi = 0; mi < 2; ++mi)
#pragma unroll
        for (int ni = 0; ni < 2; ++ni)
#pragma unroll
            for (int r = 0; r < 8; ++r) {
                int row = m0 + wm + mi * 16 + r + half * 8;
                int col = n0 + wn + ni * 16 + l16;
                float b = bias ? bias[col] : 0.0f;
                C[(size_t)row * N + col] = acc[mi][ni][r] + b;
            }
}

// ---------------------------------------------------------------------------
// NeoX RoPE applied in-place to Q [B,S,NQ,HD] and K [B,S,NKV,HD] (fp32).
// One thread per rotation pair.
// ---------------------------------------------------------------------------
__global__ __launch_bounds__(256)
void rope_kernel(float* __restrict__ qb, float* __restrict__ kb,
                 const int* __restrict__ pos) {
    int i = blockIdx.x * blockDim.x + threadIdx.x;
    const int nq_total = TOKENS * NQ * HALF;
    const int total = nq_total + TOKENS * NKV * HALF;
    if (i >= total) return;

    float* buf; int j, t, base;
    if (i < nq_total) {
        j = i % HALF; int h = (i / HALF) % NQ; t = i / (HALF * NQ);
        base = (t * NQ + h) * HD; buf = qb;
    } else {
        int i2 = i - nq_total;
        j = i2 % HALF; int h = (i2 / HALF) % NKV; t = i2 / (HALF * NKV);
        base = (t * NKV + h) * HD; buf = kb;
    }
    int s = t % SEQ;
    float p = (float)pos[s];
    float freq = p * __powf(10000.0f, -(float)(2 * j) * (1.0f / (float)HD));
    float sn, cs;
    __sincosf(freq, &sn, &cs);
    float x1 = buf[base + j], x2 = buf[base + j + HALF];
    buf[base + j]        = x1 * cs - x2 * sn;
    buf[base + j + HALF] = x2 * cs + x1 * sn;
}

// ---------------------------------------------------------------------------
// Causal flash attention (GQA 16q/4kv, HD=64), bf16 WMMA, fp32 softmax.
// Grid: (S/64, B*NQ). Block: 128 threads = 4 waves; wave owns a 16-row band.
// K tile stored [kv][d] (QK^T B-frag reads contiguous along d);
// V tile stored TRANSPOSED [d][kv] (PV B-frag reads contiguous along kv).
// ---------------------------------------------------------------------------
__global__ __launch_bounds__(128)
void flash_attn(const float* __restrict__ Q, const float* __restrict__ Kin,
                const float* __restrict__ Vin, float* __restrict__ O) {
    __shared__ __bf16 Kt[64][72];      // [kv][d]
    __shared__ __bf16 Vt[64][72];      // [d][kv]  (transposed)
    __shared__ __bf16 Pt[4][16][72];   // per-wave P tile [qrow][kv]

    const float SCALE = 0.125f;  // HD^-0.5
    const int tid = threadIdx.x, wave = tid >> 5, lane = tid & 31;
    const int half = lane >> 4, l16 = lane & 15;
    const int q0 = blockIdx.x << 6;
    const int b = blockIdx.y / NQ, h = blockIdx.y % NQ, hkv = h / (NQ / NKV);

    const float* Qp = Q   + (size_t)b * SEQ * (NQ * HD)  + h   * HD;
    const float* Kp = Kin + (size_t)b * SEQ * (NKV * HD) + hkv * HD;
    const float* Vp = Vin + (size_t)b * SEQ * (NKV * HD) + hkv * HD;

    // Resident Q fragments for this wave's 16-row band (two K-steps over HD=64)
    v16bf qa[2];
    {
        const float* qrow = Qp + (size_t)(q0 + wave * 16 + l16) * (NQ * HD);
#pragma unroll
        for (int s = 0; s < 2; ++s)
#pragma unroll
            for (int r = 0; r < 8; ++r) {
                int kb = s * 32 + frag_kbase(r, half);
                qa[s][2 * r]     = (__bf16)qrow[kb];
                qa[s][2 * r + 1] = (__bf16)qrow[kb + 1];
            }
    }

    v8f oacc[4] = {};
    float mst[8], lst[8];
#pragma unroll
    for (int r = 0; r < 8; ++r) { mst[r] = -1e30f; lst[r] = 0.0f; }

    const int ntiles = blockIdx.x + 1;   // causal: skip tiles above diagonal
    for (int t = 0; t < ntiles; ++t) {
        const int kv0 = t << 6;
        __syncthreads();
        // Cooperative tile fill: float4 global loads; K direct, V transposed.
        for (int i = tid; i < (64 * 64) / 4; i += 128) {
            int idx = i << 2;
            int r = idx >> 6, c = idx & 63;
            float4 kq = *reinterpret_cast<const float4*>(
                &Kp[(size_t)(kv0 + r) * (NKV * HD) + c]);
            st_bf16x4(&Kt[r][c], kq);
            float4 vv = *reinterpret_cast<const float4*>(
                &Vp[(size_t)(kv0 + r) * (NKV * HD) + c]);
            Vt[c + 0][r] = (__bf16)vv.x;
            Vt[c + 1][r] = (__bf16)vv.y;
            Vt[c + 2][r] = (__bf16)vv.z;
            Vt[c + 3][r] = (__bf16)vv.w;
        }
        __syncthreads();

        // S = Q @ K^T  (contraction over d; B-frag = A-layout of K[kv][d])
        v8f sacc[4] = {};
#pragma unroll
        for (int s = 0; s < 2; ++s)
#pragma unroll
            for (int nt = 0; nt < 4; ++nt) {
                v16bf bfrg;
#pragma unroll
                for (int r = 0; r < 8; ++r) {
                    int kb = s * 32 + frag_kbase(r, half);
                    bfrg[2 * r]     = Kt[nt * 16 + l16][kb];
                    bfrg[2 * r + 1] = Kt[nt * 16 + l16][kb + 1];
                }
                sacc[nt] = __builtin_amdgcn_wmma_f32_16x16x32_bf16(
                    false, qa[s], false, bfrg, (short)0, sacc[nt], false, false);
            }

        // Online softmax (rows live in a 16-lane half; reduce with shfl_xor)
        const bool diag = (t == blockIdx.x);
#pragma unroll
        for (int r = 0; r < 8; ++r) {
            int rowg = q0 + wave * 16 + r + half * 8;
            float mx = -1e30f;
#pragma unroll
            for (int nt = 0; nt < 4; ++nt) {
                float v = sacc[nt][r] * SCALE;
                if (diag && (kv0 + nt * 16 + l16) > rowg) v = -1e30f;
                sacc[nt][r] = v;
                mx = fmaxf(mx, v);
            }
            mx = fmaxf(mx, __shfl_xor(mx, 1, 32));
            mx = fmaxf(mx, __shfl_xor(mx, 2, 32));
            mx = fmaxf(mx, __shfl_xor(mx, 4, 32));
            mx = fmaxf(mx, __shfl_xor(mx, 8, 32));
            float mn = fmaxf(mst[r], mx);
            float rsum = 0.0f;
#pragma unroll
            for (int nt = 0; nt < 4; ++nt) {
                float p = __expf(sacc[nt][r] - mn);
                Pt[wave][r + half * 8][nt * 16 + l16] = (__bf16)p;
                rsum += p;
            }
            rsum += __shfl_xor(rsum, 1, 32);
            rsum += __shfl_xor(rsum, 2, 32);
            rsum += __shfl_xor(rsum, 4, 32);
            rsum += __shfl_xor(rsum, 8, 32);
            float alpha = __expf(mst[r] - mn);
            lst[r] = lst[r] * alpha + rsum;
            mst[r] = mn;
#pragma unroll
            for (int dt = 0; dt < 4; ++dt) oacc[dt][r] *= alpha;
        }
        __syncthreads();   // Pt visibility + protect Kt/Vt reuse ordering

        // O += P @ V  (contraction over kv; B-frag reads transposed V tile)
#pragma unroll
        for (int s = 0; s < 2; ++s) {
            v16bf af;
#pragma unroll
            for (int r = 0; r < 8; ++r) {
                int kb = s * 32 + frag_kbase(r, half);
                af[2 * r]     = Pt[wave][l16][kb];
                af[2 * r + 1] = Pt[wave][l16][kb + 1];
            }
#pragma unroll
            for (int dt = 0; dt < 4; ++dt) {
                v16bf bfrg;
#pragma unroll
                for (int r = 0; r < 8; ++r) {
                    int kb = s * 32 + frag_kbase(r, half);
                    bfrg[2 * r]     = Vt[dt * 16 + l16][kb];
                    bfrg[2 * r + 1] = Vt[dt * 16 + l16][kb + 1];
                }
                oacc[dt] = __builtin_amdgcn_wmma_f32_16x16x32_bf16(
                    false, af, false, bfrg, (short)0, oacc[dt], false, false);
            }
        }
    }

    // Normalize and write ctx [B,S,NQ,HD]
#pragma unroll
    for (int dt = 0; dt < 4; ++dt)
#pragma unroll
        for (int r = 0; r < 8; ++r) {
            int rowg = q0 + wave * 16 + r + half * 8;
            int d = dt * 16 + l16;
            O[(((size_t)b * SEQ + rowg) * NQ + h) * HD + d] =
                oacc[dt][r] / lst[r];
        }
}

// ---------------------------------------------------------------------------
extern "C" void kernel_launch(void* const* d_in, const int* in_sizes, int n_in,
                              void* d_out, int out_size, void* d_ws, size_t ws_size,
                              hipStream_t stream) {
    const float* hs  = (const float*)d_in[0];
    const int*   pos = (const int*)  d_in[1];
    const float* Wq  = (const float*)d_in[2];
    const float* bq  = (const float*)d_in[3];
    const float* Wk  = (const float*)d_in[4];
    const float* bk  = (const float*)d_in[5];
    const float* Wv  = (const float*)d_in[6];
    const float* bv  = (const float*)d_in[7];
    const float* Wo  = (const float*)d_in[8];
    float* out = (float*)d_out;

    float* qb  = (float*)d_ws;                       // [TOKENS, NQ*HD]
    float* kb  = qb + (size_t)TOKENS * NQ * HD;      // [TOKENS, NKV*HD]
    float* vb  = kb + (size_t)TOKENS * NKV * HD;     // [TOKENS, NKV*HD]
    float* ctx = vb + (size_t)TOKENS * NKV * HD;     // [TOKENS, NQ*HD]

    dim3 blk(128);
    gemm_bias_bf16<<<dim3((NQ * HD) / 64,  TOKENS / 64), blk, 0, stream>>>(
        hs, Wq, bq, qb, TOKENS, NQ * HD, HIDDEN);
    gemm_bias_bf16<<<dim3((NKV * HD) / 64, TOKENS / 64), blk, 0, stream>>>(
        hs, Wk, bk, kb, TOKENS, NKV * HD, HIDDEN);
    gemm_bias_bf16<<<dim3((NKV * HD) / 64, TOKENS / 64), blk, 0, stream>>>(
        hs, Wv, bv, vb, TOKENS, NKV * HD, HIDDEN);

    int rope_total = TOKENS * (NQ + NKV) * HALF;
    rope_kernel<<<(rope_total + 255) / 256, 256, 0, stream>>>(qb, kb, pos);

    flash_attn<<<dim3(SEQ / 64, BATCH * NQ), blk, 0, stream>>>(qb, kb, vb, ctx);

    gemm_bias_bf16<<<dim3(HIDDEN / 64, TOKENS / 64), blk, 0, stream>>>(
        ctx, Wo, nullptr, out, TOKENS, HIDDEN, HIDDEN);
}
